// MultiHeadSelfAttention_86260123173351
// MI455X (gfx1250) — compile-verified, matrix operand order checked by reading, not run on hardware
//
#include <hip/hip_runtime.h>

typedef __attribute__((ext_vector_type(16))) _Float16 v16h;
typedef __attribute__((ext_vector_type(8)))  float    v8f;
typedef unsigned int tdm_g0 __attribute__((ext_vector_type(4)));
typedef int          tdm_g1 __attribute__((ext_vector_type(8)));
typedef int          tdm_g2 __attribute__((ext_vector_type(4)));

#define B_  4
#define T_  2048
#define C_  1024
#define H_  16
#define HD_ 64
#define M_  (B_*T_)

#if defined(__has_builtin)
#if __has_builtin(__builtin_amdgcn_tensor_load_to_lds) && \
    __has_builtin(__builtin_amdgcn_s_wait_tensorcnt)
#define USE_TDM 1
#endif
#endif

union FragU { v16h h; uint4 u[2]; };

// A/B fragment for v_wmma_f32_16x16x32_f16.
// lane (l = lane%16, g = lane/16) holds halves:
//   h=0..7  -> k = k0 + g*8 + h        (16B contiguous)
//   h=8..15 -> k = k0 + 16 + g*8 + h-8 (16B contiguous)
__device__ __forceinline__ v16h ld_frag(const _Float16* row, int k0, int g) {
  FragU f;
  f.u[0] = *reinterpret_cast<const uint4*>(row + k0 + g * 8);
  f.u[1] = *reinterpret_cast<const uint4*>(row + k0 + 16 + g * 8);
  return f.h;
}

__device__ __forceinline__ v8f wmma_f16(v16h a, v16h b, v8f c) {
  return __builtin_amdgcn_wmma_f32_16x16x32_f16(
      false, a, false, b, (short)0, c, false, false);
}

__global__ void cvt_f32_to_f16(const float* __restrict__ in,
                               _Float16* __restrict__ out, int n) {
  int i = blockIdx.x * 256 + threadIdx.x;
  if (i < n) out[i] = (_Float16)in[i];
}

// ---------------------------------------------------------------------------
// QKV projection, 64x64 register-blocked per wave (16 WMMA per k-step).
// Q,K written [B,H,T,hd]; V written transposed [B,H,hd,T].
// ---------------------------------------------------------------------------
__global__ __launch_bounds__(128) void qkv_proj(
    const _Float16* __restrict__ xh,
    const _Float16* __restrict__ wqh, const _Float16* __restrict__ wkh,
    const _Float16* __restrict__ wvh,
    const float* __restrict__ bq, const float* __restrict__ bk,
    const float* __restrict__ bv,
    _Float16* __restrict__ Qo, _Float16* __restrict__ Ko,
    _Float16* __restrict__ Vt) {
  const int z = blockIdx.z;
  const _Float16* W   = (z == 0) ? wqh : (z == 1) ? wkh : wvh;
  const float*   bias = (z == 0) ? bq  : (z == 1) ? bk  : bv;
  const int lane = threadIdx.x & 31;
  const int wid  = threadIdx.x >> 5;
  const int l = lane & 15, g = lane >> 4;
  const int ntile = blockIdx.x * 4 + wid;   // 0..15 : 64-wide feature tile
  const int mtile = blockIdx.y;             // 0..127: 64-high token tile

  const _Float16* arow[4];
  const _Float16* brow[4];
  #pragma unroll
  for (int i = 0; i < 4; ++i) {
    arow[i] = xh + (size_t)(mtile * 64 + i * 16 + l) * C_;
    brow[i] = W  + (size_t)(ntile * 64 + i * 16 + l) * C_;
  }

  v8f acc[4][4];
  #pragma unroll
  for (int mm = 0; mm < 4; ++mm)
    #pragma unroll
    for (int nn = 0; nn < 4; ++nn) acc[mm][nn] = {};

  for (int k0 = 0; k0 < C_; k0 += 32) {
    v16h af[4], bf[4];
    #pragma unroll
    for (int i = 0; i < 4; ++i) af[i] = ld_frag(arow[i], k0, g);
    #pragma unroll
    for (int i = 0; i < 4; ++i) bf[i] = ld_frag(brow[i], k0, g);
    #pragma unroll
    for (int mm = 0; mm < 4; ++mm)
      #pragma unroll
      for (int nn = 0; nn < 4; ++nn)
        acc[mm][nn] = wmma_f16(af[mm], bf[nn], acc[mm][nn]);
  }

  const int h = ntile;                       // 64-aligned head block
  #pragma unroll
  for (int nn = 0; nn < 4; ++nn) {
    const int n = ntile * 64 + nn * 16 + l;
    const int d = nn * 16 + l;
    const float bval = bias[n];
    #pragma unroll
    for (int mm = 0; mm < 4; ++mm) {
      #pragma unroll
      for (int r = 0; r < 8; ++r) {
        const int m  = mtile * 64 + mm * 16 + r + 8 * g;
        const int b_ = m >> 11, t = m & (T_ - 1);
        const float v = acc[mm][nn][r] + bval;
        if (z == 0)
          Qo[(((size_t)(b_ * H_ + h)) * T_ + t) * HD_ + d] = (_Float16)v;
        else if (z == 1)
          Ko[(((size_t)(b_ * H_ + h)) * T_ + t) * HD_ + d] = (_Float16)v;
        else
          Vt[(((size_t)(b_ * H_ + h)) * HD_ + d) * T_ + t] = (_Float16)v;
      }
    }
  }
}

// ---------------------------------------------------------------------------
// Causal flash attention. Block = 4 waves = 4 consecutive 16-query tiles
// sharing TDM-staged K/V chunks in LDS; online softmax; P transposed
// D-layout -> A-layout through per-wave LDS scratch.
// ---------------------------------------------------------------------------
__global__ __launch_bounds__(128) void flash_attn(
    const _Float16* __restrict__ Q, const _Float16* __restrict__ K,
    const _Float16* __restrict__ Vt, _Float16* __restrict__ O) {
  __shared__ __align__(16) _Float16 Ks[32 * 64];   // [s_local][d]
  __shared__ __align__(16) _Float16 Vs[64 * 32];   // [d][s_local]
  __shared__ __align__(16) _Float16 psm[4][16 * 32];
  const int lane = threadIdx.x & 31;
  const int wid  = threadIdx.x >> 5;
  const int l = lane & 15, g = lane >> 4;
  const int qtile = blockIdx.x * 4 + wid;           // 0..127
  const int bh    = blockIdx.z * H_ + blockIdx.y;

  const _Float16* Qb = Q  + (size_t)bh * T_ * HD_;
  const _Float16* Kb = K  + (size_t)bh * T_ * HD_;
  const _Float16* Vb = Vt + (size_t)bh * HD_ * T_;  // [hd][T]
  _Float16* P = psm[wid];

  const _Float16* qrow = Qb + (size_t)(qtile * 16 + l) * HD_;
  const v16h qa0 = ld_frag(qrow, 0, g);
  const v16h qa1 = ld_frag(qrow, 32, g);

  float m_i[8], l_i[8];
  v8f o0 = {}, o1 = {}, o2 = {}, o3 = {};
  #pragma unroll
  for (int r = 0; r < 8; ++r) { m_i[r] = -__builtin_inff(); l_i[r] = 0.f; }

  const int qbase    = qtile * 16;
  const int blk_qmax = (blockIdx.x * 4 + 3) * 16 + 15;  // uniform trip count
  const float scale  = 0.125f;                           // hd^-0.5

  for (int s0 = 0; s0 <= blk_qmax; s0 += 32) {
#ifdef USE_TDM
    if (wid == 0) {
      tdm_g2 gz4 = { 0, 0, 0, 0 };
      tdm_g1 gz8 = { 0, 0, 0, 0, 0, 0, 0, 0 };
      // --- Tensor Data Mover: K chunk (32 rows x 64 halves, stride 64) ---
      const unsigned long long gk =
          (unsigned long long)(size_t)Kb + (unsigned long long)s0 * (HD_ * 2);
      tdm_g0 g0k = { 1u, (unsigned)(size_t)&Ks[0],
                     (unsigned)gk, (unsigned)(gk >> 32) | (2u << 30) };
      tdm_g1 g1k = { (int)(1u << 16),        // data_size = 2B
                     (int)(64u << 16),       // tensor_dim0 = 64
                     (int)(2048u << 16),     // tensor_dim1 = 2048
                     (int)(64u << 16),       // tile_dim0 = 64
                     32,                     // tile_dim1 = 32 rows
                     64,                     // tensor_dim0_stride = 64
                     0, 0 };
      __builtin_amdgcn_tensor_load_to_lds(g0k, g1k, gz4, gz4, gz8, 0);
      // --- TDM: V chunk (64 rows x 32 halves, stride 2048) ---
      const unsigned long long gv =
          (unsigned long long)(size_t)Vb + (unsigned long long)s0 * 2;
      tdm_g0 g0v = { 1u, (unsigned)(size_t)&Vs[0],
                     (unsigned)gv, (unsigned)(gv >> 32) | (2u << 30) };
      tdm_g1 g1v = { (int)(1u << 16),        // data_size = 2B
                     (int)(2048u << 16),     // tensor_dim0 = 2048
                     (int)(64u << 16),       // tensor_dim1 = 64
                     (int)(32u << 16),       // tile_dim0 = 32
                     64,                     // tile_dim1 = 64 rows
                     2048,                   // tensor_dim0_stride = 2048
                     0, 0 };
      __builtin_amdgcn_tensor_load_to_lds(g0v, g1v, gz4, gz4, gz8, 0);
      __builtin_amdgcn_s_wait_tensorcnt(0);
    }
#else
    {  // cooperative staging fallback: 128 threads x 2 uint4 per tile
      const int tid = threadIdx.x;
      #pragma unroll
      for (int it = 0; it < 2; ++it) {
        const int idx = tid + it * 128;          // 0..255 uint4 chunks
        const int kr = idx >> 3, kc = (idx & 7) * 8;
        *reinterpret_cast<uint4*>(&Ks[kr * 64 + kc]) =
            *reinterpret_cast<const uint4*>(Kb + (size_t)(s0 + kr) * HD_ + kc);
        const int vr = idx >> 2, vc = (idx & 3) * 8;
        *reinterpret_cast<uint4*>(&Vs[vr * 32 + vc]) =
            *reinterpret_cast<const uint4*>(Vb + (size_t)vr * T_ + s0 + vc);
      }
    }
#endif
    __syncthreads();   // K/V chunk visible to all waves

    // S = Q @ K^T (two 16-wide key subtiles), fragments from LDS
    v8f sA = {}, sB = {};
    {
      const _Float16* krow = &Ks[(0 + l) * 64];
      sA = wmma_f16(qa0, ld_frag(krow, 0, g), sA);
      sA = wmma_f16(qa1, ld_frag(krow, 32, g), sA);
    }
    {
      const _Float16* krow = &Ks[(16 + l) * 64];
      sB = wmma_f16(qa0, ld_frag(krow, 0, g), sB);
      sB = wmma_f16(qa1, ld_frag(krow, 32, g), sB);
    }

    const bool need_mask = (s0 + 31 > qbase);
    float p0[8], p1[8];
    #pragma unroll
    for (int r = 0; r < 8; ++r) {
      const int row = qbase + r + 8 * g;
      float a = sA[r] * scale, b = sB[r] * scale;
      if (need_mask) {
        if (s0 + l      > row) a = -__builtin_inff();
        if (s0 + 16 + l > row) b = -__builtin_inff();
      }
      p0[r] = a; p1[r] = b;
    }

    float alpha[8];
    #pragma unroll
    for (int r = 0; r < 8; ++r) {
      float mx = fmaxf(p0[r], p1[r]);
      #pragma unroll
      for (int off = 8; off >= 1; off >>= 1)
        mx = fmaxf(mx, __shfl_xor(mx, off, 32));      // 16-lane-group rowmax
      const float nm = fmaxf(m_i[r], mx);
      alpha[r] = __expf(m_i[r] - nm);
      m_i[r] = nm;
      const float e0 = __expf(p0[r] - nm);
      const float e1 = __expf(p1[r] - nm);
      P[(r + 8 * g) * 32 + l]      = (_Float16)e0;
      P[(r + 8 * g) * 32 + 16 + l] = (_Float16)e1;
      float rs = e0 + e1;
      #pragma unroll
      for (int off = 8; off >= 1; off >>= 1)
        rs += __shfl_xor(rs, off, 32);                // 16-lane-group rowsum
      l_i[r] = l_i[r] * alpha[r] + rs;
    }

    #pragma unroll
    for (int r = 0; r < 8; ++r) {
      o0[r] *= alpha[r]; o1[r] *= alpha[r];
      o2[r] *= alpha[r]; o3[r] *= alpha[r];
    }

    asm volatile("s_wait_dscnt 0" ::: "memory");      // P visible in LDS
    const v16h pa = ld_frag(P + l * 32, 0, g);        // A-fragment of P

    o0 = wmma_f16(pa, ld_frag(&Vs[( 0 + l) * 32], 0, g), o0);
    o1 = wmma_f16(pa, ld_frag(&Vs[(16 + l) * 32], 0, g), o1);
    o2 = wmma_f16(pa, ld_frag(&Vs[(32 + l) * 32], 0, g), o2);
    o3 = wmma_f16(pa, ld_frag(&Vs[(48 + l) * 32], 0, g), o3);

    __syncthreads();   // all waves done with this K/V chunk
  }

  #pragma unroll
  for (int r = 0; r < 8; ++r) {
    const float inv = 1.0f / l_i[r];
    const int t = qbase + r + 8 * g;
    _Float16* orow =
        O + ((size_t)blockIdx.z * T_ + t) * C_ + blockIdx.y * HD_;
    orow[ 0 + l] = (_Float16)(o0[r] * inv);
    orow[16 + l] = (_Float16)(o1[r] * inv);
    orow[32 + l] = (_Float16)(o2[r] * inv);
    orow[48 + l] = (_Float16)(o3[r] * inv);
  }
}

// ---------------------------------------------------------------------------
// Output projection, 64x64 register-blocked: out = attn @ wo^T + bo (fp32)
// ---------------------------------------------------------------------------
__global__ __launch_bounds__(128) void out_proj(
    const _Float16* __restrict__ ah, const _Float16* __restrict__ woh,
    const float* __restrict__ bo, float* __restrict__ out) {
  const int lane = threadIdx.x & 31, wid = threadIdx.x >> 5;
  const int l = lane & 15, g = lane >> 4;
  const int ntile = blockIdx.x * 4 + wid;
  const int mtile = blockIdx.y;

  const _Float16* arow[4];
  const _Float16* brow[4];
  #pragma unroll
  for (int i = 0; i < 4; ++i) {
    arow[i] = ah  + (size_t)(mtile * 64 + i * 16 + l) * C_;
    brow[i] = woh + (size_t)(ntile * 64 + i * 16 + l) * C_;
  }

  v8f acc[4][4];
  #pragma unroll
  for (int mm = 0; mm < 4; ++mm)
    #pragma unroll
    for (int nn = 0; nn < 4; ++nn) acc[mm][nn] = {};

  for (int k0 = 0; k0 < C_; k0 += 32) {
    v16h af[4], bf[4];
    #pragma unroll
    for (int i = 0; i < 4; ++i) af[i] = ld_frag(arow[i], k0, g);
    #pragma unroll
    for (int i = 0; i < 4; ++i) bf[i] = ld_frag(brow[i], k0, g);
    #pragma unroll
    for (int mm = 0; mm < 4; ++mm)
      #pragma unroll
      for (int nn = 0; nn < 4; ++nn)
        acc[mm][nn] = wmma_f16(af[mm], bf[nn], acc[mm][nn]);
  }

  #pragma unroll
  for (int nn = 0; nn < 4; ++nn) {
    const int n = ntile * 64 + nn * 16 + l;
    const float bval = bo[n];
    #pragma unroll
    for (int mm = 0; mm < 4; ++mm) {
      #pragma unroll
      for (int r = 0; r < 8; ++r) {
        const int m = mtile * 64 + mm * 16 + r + 8 * g;
        out[(size_t)m * C_ + n] = acc[mm][nn][r] + bval;
      }
    }
  }
}

extern "C" void kernel_launch(void* const* d_in, const int* in_sizes, int n_in,
                              void* d_out, int out_size, void* d_ws,
                              size_t ws_size, hipStream_t stream) {
  // inputs: x, causal_mask(unused), wq, bq, wk, bk, wv, bv, wo, bo
  const float* x  = (const float*)d_in[0];
  const float* wq = (const float*)d_in[2];
  const float* bq = (const float*)d_in[3];
  const float* wk = (const float*)d_in[4];
  const float* bk = (const float*)d_in[5];
  const float* wv = (const float*)d_in[6];
  const float* bv = (const float*)d_in[7];
  const float* wo = (const float*)d_in[8];
  const float* bo = (const float*)d_in[9];

  char* ws = (char*)d_ws;
  const size_t nx = (size_t)M_ * C_;  // 8M elems
  const size_t nw = (size_t)C_ * C_;  // 1M elems
  _Float16* xh  = (_Float16*)ws; ws += nx * 2;
  _Float16* wqh = (_Float16*)ws; ws += nw * 2;
  _Float16* wkh = (_Float16*)ws; ws += nw * 2;
  _Float16* wvh = (_Float16*)ws; ws += nw * 2;
  _Float16* woh = (_Float16*)ws; ws += nw * 2;
  _Float16* Qh  = (_Float16*)ws; ws += nx * 2;
  _Float16* Kh  = (_Float16*)ws; ws += nx * 2;
  _Float16* Vth = (_Float16*)ws; ws += nx * 2;
  _Float16* Ah  = (_Float16*)ws; ws += nx * 2;   // total ~88 MB

  cvt_f32_to_f16<<<(unsigned)((nx + 255) / 256), 256, 0, stream>>>(x,  xh,  (int)nx);
  cvt_f32_to_f16<<<(unsigned)((nw + 255) / 256), 256, 0, stream>>>(wq, wqh, (int)nw);
  cvt_f32_to_f16<<<(unsigned)((nw + 255) / 256), 256, 0, stream>>>(wk, wkh, (int)nw);
  cvt_f32_to_f16<<<(unsigned)((nw + 255) / 256), 256, 0, stream>>>(wv, wvh, (int)nw);
  cvt_f32_to_f16<<<(unsigned)((nw + 255) / 256), 256, 0, stream>>>(wo, woh, (int)nw);

  qkv_proj<<<dim3(4, 128, 3), 128, 0, stream>>>(xh, wqh, wkh, wvh,
                                                bq, bk, bv, Qh, Kh, Vth);
  flash_attn<<<dim3(32, 16, 4), 128, 0, stream>>>(Qh, Kh, Vth, Ah);
  out_proj<<<dim3(4, 128), 128, 0, stream>>>(Ah, woh, bo, (float*)d_out);
}
